// MemoryAttention_66065186947091
// MI455X (gfx1250) — compile-verified, matrix operand order checked by reading, not run on hardware
//
#include <hip/hip_runtime.h>

// ---------------------------------------------------------------------------
// Types / constants
// ---------------------------------------------------------------------------
typedef __attribute__((ext_vector_type(16))) __bf16 bf16x16;
typedef __attribute__((ext_vector_type(8)))  float  floatx8;

constexpr int kDim    = 2048;
constexpr int kNH     = 16;
constexpr int kHD     = 128;
constexpr int kMem    = 128;
constexpr int kSeq    = 2048;
constexpr int kBatch  = 4;
constexpr int kHid    = 5632;
constexpr int kChunks = kSeq / kMem;     // 16
constexpr int kRows   = kBatch * kMem;   // 512 rows per step GEMM
constexpr int kTot    = 2 * kMem;        // 256 attention length
constexpr float kEps  = 1e-5f;

// ---------------------------------------------------------------------------
// CDNA5 primitives
// ---------------------------------------------------------------------------
__device__ inline floatx8 wmma_bf16(bf16x16 a, bf16x16 b, floatx8 c) {
  return __builtin_amdgcn_wmma_f32_16x16x32_bf16(false, a, false, b, (short)0,
                                                 c, false, false);
}

// Async global->LDS 16B copy per lane (ASYNCcnt tracked, no VGPR round trip).
__device__ inline void async_ld16(void* lds, const void* gaddr) {
  unsigned l = (unsigned)(uintptr_t)lds;  // low 32 bits of generic = LDS offset
  asm volatile("global_load_async_to_lds_b128 %0, %1, off"
               :: "v"(l), "v"(gaddr) : "memory");
}
__device__ inline void wait_async() {
  asm volatile("s_wait_asynccnt 0x0" ::: "memory");
}

// Unified fragment load: per-lane row r = lane&15, K-halves split by lane>>4
// (ISA 7.12.2 for both A (r=M) and B (r=N) operands). Source must be
// "r-major" with 16B-aligned rows: ld multiple of 8 elements.
// Lowers to exactly two ds_load_b128.
__device__ inline bf16x16 load_frag16(const __bf16* p, int ld) {
  const int lane = threadIdx.x & 31;
  const int r = lane & 15;
  const int kh = (lane >> 4) << 3;
  union { uint4 u[2]; bf16x16 v; } t;
  t.u[0] = *(const uint4*)(p + r * ld + kh);
  t.u[1] = *(const uint4*)(p + r * ld + 16 + kh);
  return t.v;
}

// C/D 16x16 f32 layout: VGPR j -> M = (lanehalf*8 + j), N = lane&15.
__device__ inline void store_c_f32(float* p, int ldc, floatx8 acc) {
  const int lane = threadIdx.x & 31;
  const int n = lane & 15;
  const int mh = (lane >> 4) << 3;
#pragma unroll
  for (int j = 0; j < 8; ++j) p[(mh + j) * ldc + n] = acc[j];
}

// ---------------------------------------------------------------------------
// GEMM: C[M,N](f32) = A[M,K](bf16 row major) * Wt[N,K](bf16 pre-transposed)
// 256 threads (8 waves); C tile 128x128; wave tile 32x64 (8 WMMA accums);
// K-block 32, async double-buffered LDS (one barrier per K-block).
// blockIdx.z selects among up to 3 (Wt, C) pairs sharing the same A
// (batches independent projections into a single launch).
// ---------------------------------------------------------------------------
__global__ __launch_bounds__(256) void gemm_bf16_kernel(
    const __bf16* __restrict__ A,
    const __bf16* Bt0, const __bf16* Bt1, const __bf16* Bt2,
    float* C0, float* C1, float* C2,
    int N, int K, long long aTileStride) {
  constexpr int TM = 128, TN = 128, KB = 32, LD = KB + 8;  // 40 elems = 80B
  __shared__ __attribute__((aligned(16))) __bf16 As[2][TM * LD];  // 2x10KB
  __shared__ __attribute__((aligned(16))) __bf16 Bs[2][TN * LD];  // 2x10KB

  const int tid = threadIdx.x;
  const __bf16* Bt = (blockIdx.z == 0) ? Bt0 : (blockIdx.z == 1 ? Bt1 : Bt2);
  float* C = (blockIdx.z == 0) ? C0 : (blockIdx.z == 1 ? C1 : C2);
  const __bf16* Ab = A + (long long)blockIdx.y * aTileStride;
  const __bf16* Bb = Bt + (long long)blockIdx.x * TN * K;  // N-major rows
  const int w = tid >> 5, wm = w >> 1, wn = w & 1;

  auto stage = [&](int buf, int kb) {
#pragma unroll
    for (int i = 0; i < 2; ++i) {  // A and B: 128 rows x 32 each
      int v = tid + i * 256;
      int row = v >> 2, seg = v & 3;
      async_ld16(&As[buf][row * LD + seg * 8],
                 Ab + (long long)row * K + kb + seg * 8);
      async_ld16(&Bs[buf][row * LD + seg * 8],
                 Bb + (long long)row * K + kb + seg * 8);
    }
  };

  floatx8 acc[2][4] = {};
  stage(0, 0);
  int buf = 0;
  for (int kb = 0; kb < K; kb += KB) {
    wait_async();
    __syncthreads();                           // current buf ready everywhere;
    if (kb + KB < K) stage(buf ^ 1, kb + KB);  // prior reads of buf^1 done
    bf16x16 a0 = load_frag16(&As[buf][(wm * 32) * LD], LD);
    bf16x16 a1 = load_frag16(&As[buf][(wm * 32 + 16) * LD], LD);
#pragma unroll
    for (int j = 0; j < 4; ++j) {
      bf16x16 bj = load_frag16(&Bs[buf][(wn * 64 + j * 16) * LD], LD);
      acc[0][j] = wmma_bf16(a0, bj, acc[0][j]);
      acc[1][j] = wmma_bf16(a1, bj, acc[1][j]);
    }
    buf ^= 1;
  }

  float* Cb = C + (long long)blockIdx.y * TM * N + blockIdx.x * TN;
#pragma unroll
  for (int i = 0; i < 2; ++i)
#pragma unroll
    for (int j = 0; j < 4; ++j)
      store_c_f32(Cb + (wm * 32 + i * 16) * N + wn * 64 + j * 16, N, acc[i][j]);
}

// ---------------------------------------------------------------------------
// Weight transpose + bf16 convert: Wt[n][k] = (bf16)W[k][n]. 32x32 tiles.
// ---------------------------------------------------------------------------
__global__ __launch_bounds__(256) void transpose_cvt_kernel(
    const float* __restrict__ W, __bf16* __restrict__ Wt, int K, int N) {
  __shared__ float tile[32][33];
  const int n0 = blockIdx.x * 32, k0 = blockIdx.y * 32;
  const int tx = threadIdx.x & 31, ty = threadIdx.x >> 5;  // ty in 0..7
#pragma unroll
  for (int i = 0; i < 4; ++i)
    tile[ty + i * 8][tx] = W[(long long)(k0 + ty + i * 8) * N + n0 + tx];
  __syncthreads();
#pragma unroll
  for (int i = 0; i < 4; ++i)
    Wt[(long long)(n0 + ty + i * 8) * K + k0 + tx] =
        (__bf16)tile[tx][ty + i * 8];
}

// ---------------------------------------------------------------------------
// Elementwise kernels
// ---------------------------------------------------------------------------
__global__ void cvt_f32_bf16_kernel(const float* __restrict__ in,
                                    __bf16* __restrict__ out, long long n) {
  long long i = (long long)blockIdx.x * blockDim.x + threadIdx.x;
  if (i < n) out[i] = (__bf16)in[i];
}

__global__ __launch_bounds__(256) void add_rmsnorm_kernel(
    const float* __restrict__ a, const float* __restrict__ b,
    const float* __restrict__ w, __bf16* __restrict__ out) {
  const long long row = blockIdx.x;
  const float* ar = a + row * kDim;
  const float* br = b ? (b + row * kDim) : nullptr;
  float vals[8];
  float ss = 0.f;
#pragma unroll
  for (int i = 0; i < 8; ++i) {
    int c = threadIdx.x + i * 256;
    float v = ar[c] + (br ? br[c] : 0.f);
    vals[i] = v;
    ss += v * v;
  }
  __shared__ float red[256];
  red[threadIdx.x] = ss;
  __syncthreads();
  for (int s = 128; s > 0; s >>= 1) {
    if (threadIdx.x < s) red[threadIdx.x] += red[threadIdx.x + s];
    __syncthreads();
  }
  float scale = rsqrtf(red[0] / (float)kDim + kEps);
#pragma unroll
  for (int i = 0; i < 8; ++i) {
    int c = threadIdx.x + i * 256;
    out[row * kDim + c] = (__bf16)(vals[i] * scale * w[c]);
  }
}

__global__ void silu_mul_kernel(const float* __restrict__ u,
                                const float* __restrict__ g,
                                __bf16* __restrict__ out, long long n) {
  long long i = (long long)blockIdx.x * blockDim.x + threadIdx.x;
  if (i < n) {
    float x = u[i];
    out[i] = (__bf16)((x / (1.f + __expf(-x))) * g[i]);
  }
}

// Q/K packed (b,h,t,d) with RoPE; V packed (b,h,d,t) so that P@V B-fragments
// are per-lane contiguous.  mq == mk per the reference.
__global__ __launch_bounds__(128) void rope_pack_kernel(
    const float* __restrict__ mk, const float* __restrict__ mv,
    const float* __restrict__ xq, const float* __restrict__ xk,
    const float* __restrict__ xv, const float* __restrict__ fcos,
    const float* __restrict__ fsin, __bf16* __restrict__ qh,
    __bf16* __restrict__ kh, __bf16* __restrict__ vh) {
  const int d = threadIdx.x;
  const int hh = blockIdx.x & (kNH - 1);
  const int t = (blockIdx.x >> 4) & (kTot - 1);
  const int b = blockIdx.x >> 12;

  const float *qs, *ks, *vs;
  if (t < kMem) {
    long long ro = ((long long)(b * kMem + t)) * kDim + hh * kHD;
    qs = mk + ro; ks = mk + ro; vs = mv + ro;
  } else {
    long long ro = ((long long)(b * kMem + (t - kMem))) * kDim + hh * kHD;
    qs = xq + ro; ks = xk + ro; vs = xv + ro;
  }
  const int j = d >> 1;
  const float c = fcos[t * (kHD / 2) + j];
  const float s = fsin[t * (kHD / 2) + j];
  const int pe = d & ~1;
  float qr = qs[pe], qi = qs[pe + 1];
  float kr = ks[pe], ki = ks[pe + 1];
  float qv = (d & 1) ? (qr * s + qi * c) : (qr * c - qi * s);
  float kv = (d & 1) ? (kr * s + ki * c) : (kr * c - ki * s);
  long long bh = (long long)(b * kNH + hh);
  long long dqk = (bh * kTot + t) * kHD + d;
  long long dv = (bh * kHD + d) * kTot + t;
  qh[dqk] = (__bf16)qv;
  kh[dqk] = (__bf16)kv;
  vh[dv] = (__bf16)vs[d];
}

// ---------------------------------------------------------------------------
// Attention: WG per (b, head, 64-query block of the upper 128 rows).
// 128 threads = 4 waves, each wave owns 16 query rows. K/V tiles double-
// buffered through LDS with async loads; scores in registers; causal register
// softmax via shfl; P stored bf16 in LDS; P@V WMMA; bf16 result into ys.
// ---------------------------------------------------------------------------
constexpr int kLQ = 136;   // Q LDS row stride (elems)
constexpr int kLK = 136;   // K tile row stride (t-major rows of 128 d)
constexpr int kLV = 72;    // V tile row stride (d-major rows of 64 t)
constexpr int kLP = 264;   // P row stride (256 cols)
constexpr int kKVElems = 128 * kLV;  // 9216 >= 64*kLK (8704)

__global__ __launch_bounds__(128) void attn_kernel(
    const __bf16* __restrict__ qg, const __bf16* __restrict__ kg,
    const __bf16* __restrict__ vg, __bf16* __restrict__ ysb) {
  __shared__ __attribute__((aligned(16))) __bf16 Qs[64 * kLQ];      // 17.0KB
  __shared__ __attribute__((aligned(16))) __bf16 KVs[2][kKVElems];  // 36.9KB
  __shared__ __attribute__((aligned(16))) __bf16 Pb[64 * kLP];      // 33.8KB

  const int tid = threadIdx.x;
  const int wave = tid >> 5;
  const int lane = tid & 31;
  const int idx = blockIdx.x;
  const int qb = idx & 1;
  const int hh = (idx >> 1) & (kNH - 1);
  const int b = idx >> 5;
  const long long base = ((long long)(b * kNH + hh)) * kTot * kHD;
  const __bf16* qgb = qg + base;
  const __bf16* kgb = kg + base;
  const __bf16* vgb = vg + base;  // (d, t) layout
  const float scale = 0.08838834764831845f;  // 1/sqrt(128)

  auto stage_k = [&](int buf, int kt) {  // 32 t-rows x 128 d
#pragma unroll
    for (int i = 0; i < 4; ++i) {
      int v = tid + i * 128;
      int row = v >> 4, seg = v & 15;
      async_ld16(&KVs[buf][row * kLK + seg * 8],
                 kgb + (long long)(kt * 64 + row) * kHD + seg * 8);
    }
  };
  auto stage_v = [&](int buf, int vt) {  // 64 d-rows x 64 t
#pragma unroll
    for (int i = 0; i < 4; ++i) {
      int v = tid + i * 128;
      int row = v >> 3, seg = v & 7;
      async_ld16(&KVs[buf][row * kLV + seg * 8],
                 vgb + (long long)row * kTot + vt * 64 + seg * 8);
    }
  };

  // Issue Q rows (128 + qb*64 ..) and first K tile.
#pragma unroll
  for (int i = 0; i < 4; ++i) {
    int v = tid + i * 128;
    int row = v >> 4, seg = v & 15;
    async_ld16(&Qs[row * kLQ + seg * 8],
               qgb + (long long)(kMem + qb * 64 + row) * kHD + seg * 8);
  }
  stage_k(0, 0);

  floatx8 s[16] = {};  // 16 col-tiles of 16x16 scores per wave
  for (int kt = 0; kt < 4; ++kt) {
    wait_async();
    __syncthreads();
    if (kt < 3) stage_k((kt + 1) & 1, kt + 1);
    else        stage_v(0, 0);  // overlap V0 transfer with softmax
    const __bf16* kb = KVs[kt & 1];
#pragma unroll
    for (int kk = 0; kk < 4; ++kk) {
      bf16x16 a = load_frag16(Qs + (wave * 16) * kLQ + kk * 32, kLQ);
#pragma unroll
      for (int ct = 0; ct < 4; ++ct) {
        bf16x16 bt = load_frag16(kb + (ct * 16) * kLK + kk * 32, kLK);
        s[kt * 4 + ct] = wmma_bf16(a, bt, s[kt * 4 + ct]);
      }
    }
  }

  // Register softmax (causal); row m lives in VGPR j across one 16-lane half.
#pragma unroll
  for (int j = 0; j < 8; ++j) {
    const int m = ((lane >> 4) << 3) + j;
    const int qrow = qb * 64 + wave * 16 + m;  // chunk-local output row
    const int lim = kMem + qrow;               // last valid key column
    const int nb = lane & 15;
    float mx = -3.0e38f;
#pragma unroll
    for (int ct = 0; ct < 16; ++ct) {
      int col = ct * 16 + nb;
      if (col <= lim) mx = fmaxf(mx, s[ct][j]);
    }
#pragma unroll
    for (int d = 1; d < 16; d <<= 1) mx = fmaxf(mx, __shfl_xor(mx, d, 32));
    float sum = 0.f;
    float p[16];
#pragma unroll
    for (int ct = 0; ct < 16; ++ct) {
      int col = ct * 16 + nb;
      float e = (col <= lim) ? __expf((s[ct][j] - mx) * scale) : 0.f;
      p[ct] = e;
      sum += e;
    }
#pragma unroll
    for (int d = 1; d < 16; d <<= 1) sum += __shfl_xor(sum, d, 32);
    float inv = 1.f / sum;
#pragma unroll
    for (int ct = 0; ct < 16; ++ct)
      Pb[(wave * 16 + m) * kLP + ct * 16 + nb] = (__bf16)(p[ct] * inv);
  }

  // Out = P (64x256) @ V (256x128), V tiles double-buffered.
  floatx8 o[8] = {};
  for (int vt = 0; vt < 4; ++vt) {
    wait_async();
    __syncthreads();
    if (vt < 3) stage_v((vt + 1) & 1, vt + 1);
    const __bf16* vb = KVs[vt & 1];
#pragma unroll
    for (int kk = 0; kk < 2; ++kk) {
      bf16x16 a = load_frag16(Pb + (wave * 16) * kLP + vt * 64 + kk * 32, kLP);
#pragma unroll
      for (int nt = 0; nt < 8; ++nt) {
        bf16x16 bb = load_frag16(vb + (nt * 16) * kLV + kk * 32, kLV);
        o[nt] = wmma_bf16(a, bb, o[nt]);
      }
    }
  }

  // Store om_new rows straight into ys (bf16, chunk base already applied).
  const int n = lane & 15;
  const int mh = (lane >> 4) << 3;
#pragma unroll
  for (int nt = 0; nt < 8; ++nt)
#pragma unroll
    for (int j = 0; j < 8; ++j) {
      int t_local = qb * 64 + wave * 16 + mh + j;
      long long di = (long long)b * kSeq * kDim + (long long)t_local * kDim +
                     hh * kHD + nt * 16 + n;
      ysb[di] = (__bf16)o[nt][j];
    }
}

// ---------------------------------------------------------------------------
// Host orchestration
// ---------------------------------------------------------------------------
extern "C" void kernel_launch(void* const* d_in, const int* in_sizes, int n_in,
                              void* d_out, int out_size, void* d_ws,
                              size_t ws_size, hipStream_t stream) {
  const float* x          = (const float*)d_in[0];
  const float* fcos       = (const float*)d_in[1];
  const float* fsin       = (const float*)d_in[2];
  const float* origin_mem = (const float*)d_in[3];
  const float* wq         = (const float*)d_in[4];
  const float* wk         = (const float*)d_in[5];
  const float* wv         = (const float*)d_in[6];
  const float* wo         = (const float*)d_in[7];
  // d_in[8] = wqm: unused by the reference (mq == mk)
  const float* wkm        = (const float*)d_in[9];
  const float* wvm        = (const float*)d_in[10];
  const float* wm         = (const float*)d_in[11];
  const float* ffn_norm_w = (const float*)d_in[12];
  const float* w1         = (const float*)d_in[13];
  const float* w2         = (const float*)d_in[14];
  const float* w3         = (const float*)d_in[15];
  const float* mem_norm_w = (const float*)d_in[16];

  char* ws = (char*)d_ws;
  size_t off = 0;
  auto alloc = [&](size_t bytes) -> void* {
    size_t a = (off + 255) & ~(size_t)255;
    off = a + bytes;
    return (void*)(ws + a);
  };

  const long long DD = (long long)kDim * kDim;
  const long long DH = (long long)kDim * kHid;
  const long long NX = (long long)kBatch * kSeq * kDim;
  const long long NQ = (long long)kBatch * kNH * kTot * kHD;

  // Transposed bf16 weights (~125 MB total -> L2 resident across the scan)
  __bf16* wq_t  = (__bf16*)alloc(DD * 2);
  __bf16* wk_t  = (__bf16*)alloc(DD * 2);
  __bf16* wv_t  = (__bf16*)alloc(DD * 2);
  __bf16* wo_t  = (__bf16*)alloc(DD * 2);
  __bf16* wkm_t = (__bf16*)alloc(DD * 2);
  __bf16* wvm_t = (__bf16*)alloc(DD * 2);
  __bf16* wm_t  = (__bf16*)alloc(DD * 2);
  __bf16* w1_t  = (__bf16*)alloc(DH * 2);
  __bf16* w2_t  = (__bf16*)alloc(DH * 2);
  __bf16* w3_t  = (__bf16*)alloc(DH * 2);
  // activations
  __bf16* x_b   = (__bf16*)alloc(NX * 2);
  __bf16* ys_b  = (__bf16*)alloc(NX * 2);  // scan outputs (also next-step om)
  __bf16* om0_b = (__bf16*)alloc((long long)kRows * kDim * 2);
  float*  om1   = (float*)alloc((long long)kRows * kDim * 4);
  float*  ff    = (float*)alloc((long long)kRows * kDim * 4);
  __bf16* h_b   = (__bf16*)alloc((long long)kRows * kDim * 2);
  __bf16* om3_b = (__bf16*)alloc((long long)kRows * kDim * 2);
  float*  u     = (float*)alloc((long long)kRows * kHid * 4);
  float*  g     = (float*)alloc((long long)kRows * kHid * 4);
  __bf16* t_b   = (__bf16*)alloc((long long)kRows * kHid * 2);
  float*  mk    = (float*)alloc((long long)kRows * kDim * 4);
  float*  mv    = (float*)alloc((long long)kRows * kDim * 4);
  float*  xqf   = (float*)alloc((long long)kRows * kDim * 4);
  float*  xkf   = (float*)alloc((long long)kRows * kDim * 4);
  float*  xvf   = (float*)alloc((long long)kRows * kDim * 4);
  __bf16* qh    = (__bf16*)alloc(NQ * 2);
  __bf16* kh    = (__bf16*)alloc(NQ * 2);
  __bf16* vh    = (__bf16*)alloc(NQ * 2);
  (void)ws_size; (void)in_sizes; (void)n_in; (void)out_size;

  auto tcvt = [&](const float* W, __bf16* Wt, int K, int N) {
    transpose_cvt_kernel<<<dim3(N / 32, K / 32), 256, 0, stream>>>(W, Wt, K, N);
  };
  tcvt(wq, wq_t, kDim, kDim);   tcvt(wk, wk_t, kDim, kDim);
  tcvt(wv, wv_t, kDim, kDim);   tcvt(wo, wo_t, kDim, kDim);
  tcvt(wkm, wkm_t, kDim, kDim); tcvt(wvm, wvm_t, kDim, kDim);
  tcvt(wm, wm_t, kDim, kDim);
  tcvt(w1, w1_t, kDim, kHid);   tcvt(w3, w3_t, kDim, kHid);
  tcvt(w2, w2_t, kHid, kDim);

  auto cvt = [&](const float* src, __bf16* dst, long long n) {
    cvt_f32_bf16_kernel<<<(unsigned)((n + 255) / 256), 256, 0, stream>>>(src, dst, n);
  };
  cvt(x, x_b, NX);
  cvt(origin_mem, om0_b, (long long)kRows * kDim);

  // Up to 3 independent (Wt, C) pairs sharing one A, batched via blockIdx.z.
  auto gemmN = [&](const __bf16* A, const __bf16* B0, const __bf16* B1,
                   const __bf16* B2, float* C0, float* C1, float* C2, int nB,
                   int M, int N, int K, long long aTS) {
    gemm_bf16_kernel<<<dim3(N / 128, M / 128, nB), 256, 0, stream>>>(
        A, B0, B1, B2, C0, C1, C2, N, K, aTS);
  };
  auto gemm1 = [&](const __bf16* A, const __bf16* B0, float* C0, int M, int N,
                   int K, long long aTS) {
    gemmN(A, B0, B0, B0, C0, C0, C0, 1, M, N, K, aTS);
  };

  const long long xTS = (long long)kSeq * kDim;  // batch stride in (B,SEQ,D)
  const long long cTS = 128LL * kDim;            // contiguous 512-row buffers
  const long long hTS = 128LL * kHid;

  for (int c = 0; c < kChunks; ++c) {
    // om = om @ wm   (om from origin_mem or previous chunk's ys rows)
    const __bf16* omb;
    long long omTS;
    if (c == 0) { omb = om0_b; omTS = cTS; }
    else        { omb = ys_b + (long long)(c - 1) * kMem * kDim; omTS = xTS; }
    gemm1(omb, wm_t, om1, kRows, kDim, kDim, omTS);

    // h = rmsnorm(om1); FFN (u and g batched into one launch)
    add_rmsnorm_kernel<<<kRows, 256, 0, stream>>>(om1, nullptr, ffn_norm_w, h_b);
    gemmN(h_b, w1_t, w3_t, w3_t, u, g, g, 2, kRows, kHid, kDim, cTS);
    silu_mul_kernel<<<(unsigned)(((long long)kRows * kHid) / 256), 256, 0,
                      stream>>>(u, g, t_b, (long long)kRows * kHid);
    gemm1(t_b, w2_t, ff, kRows, kDim, kHid, hTS);
    add_rmsnorm_kernel<<<kRows, 256, 0, stream>>>(om1, ff, mem_norm_w, om3_b);

    // projections: {mk,mv} batched; {xq,xk,xv} batched
    gemmN(om3_b, wkm_t, wvm_t, wvm_t, mk, mv, mv, 2, kRows, kDim, kDim, cTS);
    const __bf16* xcb = x_b + (long long)c * kMem * kDim;
    gemmN(xcb, wq_t, wk_t, wv_t, xqf, xkf, xvf, 3, kRows, kDim, kDim, xTS);

    // RoPE + head-major packing, then attention -> ys (bf16)
    rope_pack_kernel<<<kBatch * kTot * kNH, 128, 0, stream>>>(
        mk, mv, xqf, xkf, xvf, fcos, fsin, qh, kh, vh);
    attn_kernel<<<kBatch * kNH * 2, 128, 0, stream>>>(
        qh, kh, vh, ys_b + (long long)c * kMem * kDim);
  }

  // Final projection: out = ys @ wo  (8192 x 2048 x 2048, fp32 output)
  gemm1(ys_b, wo_t, (float*)d_out, kBatch * kSeq, kDim, kDim, cTS);
}
